// RNNAgent_61134564491522
// MI455X (gfx1250) — matrix operand order repaired and checked by reading, not compiled, not run on hardware
//
#include <hip/hip_runtime.h>
#include <hip/hip_bf16.h>

typedef __bf16 bf16_t;
typedef __attribute__((ext_vector_type(16))) __bf16 v16bf;
typedef __attribute__((ext_vector_type(8)))  float  v8f;

#define NB   4096
#define NA   32
#define DIN  96
#define HD   128
#define NACT 16
#define NEGC 9000000000000000.0f

// workspace offsets (in bf16 halves); each B-frag block = 32 lanes * 16 halves = 512
#define WS_FC1 0        // 8 coltiles * 3 ksteps
#define WS_WIH 12288    // 24 * 4
#define WS_WHH 61440    // 24 * 4
#define WS_A1V 110592   // 8 * 4
#define WS_A1K 126976
#define WS_A1Q 143360
#define WS_A2V 159744
#define WS_A2K 176128
#define WS_A2Q 192512
#define WS_FC2 208896   // 1 * 4

// ---------------- weight packing: W[Hout][K] -> B-fragment-major bf16 ----------------
// B-frag element j of lane l for tile (ct,ks):  B[k][n] = W[n][k],
//   n = ct*16 + (l&15),  k = ks*32 + (l&16) + j     (contiguous k: lanes0-15 K=0..15, lanes16-31 K=16..31, same as A)
__global__ void pack_w(const float* __restrict__ W, bf16_t* __restrict__ dst,
                       int coltiles, int ksteps, int K) {
    int idx = blockIdx.x * 256 + threadIdx.x;
    int total = coltiles * ksteps * 512;
    if (idx >= total) return;
    int j    = idx & 15;
    int lane = (idx >> 4) & 31;
    int blk  = idx >> 9;
    int ks   = blk % ksteps;
    int ct   = blk / ksteps;
    int n = ct * 16 + (lane & 15);
    int k = ks * 32 + (lane & 16) + j;
    dst[idx] = (bf16_t)W[n * K + k];
}

// ---------------- fragment builders ----------------
// A 16x32 bf16 layout: lane row = m0+(lane&15); lanes<16 -> K 0..15, lanes>=16 -> K 16..31
__device__ __forceinline__ v16bf load_a_bf(const bf16_t* src, int stride, int m0, int k0, int lane) {
    const bf16_t* p = src + (m0 + (lane & 15)) * stride + k0 + (lane & 16);
    v16bf f;
#pragma unroll
    for (int j = 0; j < 8; ++j) f[j] = p[j];
#pragma unroll
    for (int j = 0; j < 8; ++j) f[j + 8] = p[j + 8];
    return f;
}

__device__ __forceinline__ v16bf load_a_f32(const float* src, int stride, int m0, int lane) {
    const float* p = src + (m0 + (lane & 15)) * stride + (lane & 16);
    v16bf f;
#pragma unroll
    for (int j = 0; j < 8; ++j) f[j] = (bf16_t)p[j];
#pragma unroll
    for (int j = 0; j < 8; ++j) f[j + 8] = (bf16_t)p[j + 8];
    return f;
}

__device__ __forceinline__ v16bf load_b_ws(const bf16_t* ws, int base, int ksteps,
                                           int ct, int ks, int lane) {
    return *(const v16bf*)(ws + base + (size_t)((ct * ksteps + ks) * 32 + lane) * 16);
}

__device__ __forceinline__ v8f wmma_bf(v16bf a, v16bf b, v8f c) {
    return __builtin_amdgcn_wmma_f32_16x16x32_bf16(false, a, false, b, (short)0, c, false, false);
}

__device__ __forceinline__ float sigm(float x) { return 1.f / (1.f + __expf(-x)); }

// ---------------- fused agent kernel: one workgroup per batch element ----------------
__global__ __launch_bounds__(256) void fused_agent(
    const float* __restrict__ inp, const float* __restrict__ h0,
    const float* __restrict__ masks,
    const float* __restrict__ fc1_b, const float* __restrict__ b_ih,
    const float* __restrict__ b_hh,  const float* __restrict__ fc2_b,
    const float* __restrict__ a1v_b, const float* __restrict__ a1k_b, const float* __restrict__ a1q_b,
    const float* __restrict__ a2v_b, const float* __restrict__ a2k_b, const float* __restrict__ a2q_b,
    const bf16_t* __restrict__ ws, float* __restrict__ out) {
    __shared__ __align__(32) unsigned char smem[61440];
    float*  s_h   = (float*)(smem);             // 32x128 f32
    bf16_t* s_hb  = (bf16_t*)(smem + 16384);    // 32x128 bf16
    bf16_t* s_x   = (bf16_t*)(smem + 24576);    // 32x128 bf16 (fc1 out)
    float*  s_att = (float*)(smem + 32768);     // 32x32  f32
    bf16_t* s_xin = (bf16_t*)(smem + 36864);    // 32x96  (aliased with s_v)
    bf16_t* s_v   = (bf16_t*)(smem + 36864);    // 32x128
    bf16_t* s_k   = (bf16_t*)(smem + 45056);    // 32x128
    bf16_t* s_q   = (bf16_t*)(smem + 53248);    // 32x128

    const int tid  = threadIdx.x;
    const int lane = tid & 31;
    const int wav  = tid >> 5;
    const int b    = blockIdx.x;
    const size_t rowbase = (size_t)b * NA;

    // ---- stage inputs ----
    for (int i = tid; i < NA * DIN; i += 256) s_xin[i] = (bf16_t)inp[rowbase * DIN + i];
    for (int i = tid; i < NA * HD; i += 256) {
        float v = h0[rowbase * HD + i];
        s_h[i] = v; s_hb[i] = (bf16_t)v;
    }
    __syncthreads();

    // ---- fc1 + relu -> s_x ----
    for (int t = wav; t < 16; t += 8) {
        int m0 = (t & 1) * 16, n0 = (t >> 1) * 16;
        v8f acc = {};
#pragma unroll
        for (int ks = 0; ks < 3; ++ks)
            acc = wmma_bf(load_a_bf(s_xin, DIN, m0, ks * 32, lane),
                          load_b_ws(ws, WS_FC1, 3, n0 >> 4, ks, lane), acc);
        int col = n0 + (lane & 15);
        float bias = fc1_b[col];
        int r0 = m0 + ((lane & 16) >> 1);
#pragma unroll
        for (int r = 0; r < 8; ++r) {
            float v = acc[r] + bias;
            s_x[(r0 + r) * HD + col] = (bf16_t)(v > 0.f ? v : 0.f);
        }
    }
    __syncthreads();

    // ---- GRU: h = (1-z)*n + z*h ----
    float hnew[2][8]; int hr0[2], hcl[2];
#pragma unroll
    for (int it = 0; it < 2; ++it) {
        int t = wav + it * 8;
        int m0 = (t & 1) * 16, c0 = (t >> 1) * 16;
        v8f air = {}, aiz = {}, ain = {}, ahr = {}, ahz = {}, ahn = {};
        int ct_r = c0 >> 4, ct_z = (c0 + 128) >> 4, ct_n = (c0 + 256) >> 4;
#pragma unroll
        for (int ks = 0; ks < 4; ++ks) {
            v16bf ax = load_a_bf(s_x,  HD, m0, ks * 32, lane);
            v16bf ah = load_a_bf(s_hb, HD, m0, ks * 32, lane);
            air = wmma_bf(ax, load_b_ws(ws, WS_WIH, 4, ct_r, ks, lane), air);
            aiz = wmma_bf(ax, load_b_ws(ws, WS_WIH, 4, ct_z, ks, lane), aiz);
            ain = wmma_bf(ax, load_b_ws(ws, WS_WIH, 4, ct_n, ks, lane), ain);
            ahr = wmma_bf(ah, load_b_ws(ws, WS_WHH, 4, ct_r, ks, lane), ahr);
            ahz = wmma_bf(ah, load_b_ws(ws, WS_WHH, 4, ct_z, ks, lane), ahz);
            ahn = wmma_bf(ah, load_b_ws(ws, WS_WHH, 4, ct_n, ks, lane), ahn);
        }
        int col = c0 + (lane & 15);
        float bir = b_ih[col], biz = b_ih[col + 128], bin_ = b_ih[col + 256];
        float bhr = b_hh[col], bhz = b_hh[col + 128], bhn = b_hh[col + 256];
        int r0 = m0 + ((lane & 16) >> 1);
        hr0[it] = r0; hcl[it] = col;
#pragma unroll
        for (int r = 0; r < 8; ++r) {
            float rr = sigm(air[r] + bir + ahr[r] + bhr);
            float zz = sigm(aiz[r] + biz + ahz[r] + bhz);
            float nn = tanhf(ain[r] + bin_ + rr * (ahn[r] + bhn));
            float hold = s_h[(r0 + r) * HD + col];
            hnew[it][r] = (1.f - zz) * nn + zz * hold;
        }
    }
    __syncthreads();   // all reads of s_hb/s_h/s_x complete
#pragma unroll
    for (int it = 0; it < 2; ++it)
#pragma unroll
        for (int r = 0; r < 8; ++r) {
            int o = (hr0[it] + r) * HD + hcl[it];
            s_h[o] = hnew[it][r]; s_hb[o] = (bf16_t)hnew[it][r];
        }
    __syncthreads();

    // ---- two attention layers ----
    for (int layer = 0; layer < 2; ++layer) {
        int wbase[3];
        const float* bias3[3];
        if (layer == 0) { wbase[0]=WS_A1V; wbase[1]=WS_A1K; wbase[2]=WS_A1Q;
                          bias3[0]=a1v_b; bias3[1]=a1k_b; bias3[2]=a1q_b; }
        else            { wbase[0]=WS_A2V; wbase[1]=WS_A2K; wbase[2]=WS_A2Q;
                          bias3[0]=a2v_b; bias3[1]=a2k_b; bias3[2]=a2q_b; }
        // v,k,q projections (48 tiles)
        for (int t = wav; t < 48; t += 8) {
            int mat = t / 16, tt = t % 16;
            int m0 = (tt & 1) * 16, n0 = (tt >> 1) * 16;
            bf16_t* dst = (mat == 0) ? s_v : (mat == 1) ? s_k : s_q;
            v8f acc = {};
#pragma unroll
            for (int ks = 0; ks < 4; ++ks)
                acc = wmma_bf(load_a_bf(s_hb, HD, m0, ks * 32, lane),
                              load_b_ws(ws, wbase[mat], 4, n0 >> 4, ks, lane), acc);
            int col = n0 + (lane & 15);
            float bias = bias3[mat][col];
            int r0 = m0 + ((lane & 16) >> 1);
#pragma unroll
            for (int r = 0; r < 8; ++r) {
                float v = acc[r] + bias;
                dst[(r0 + r) * HD + col] = (bf16_t)(v > 0.f ? v : 0.f);
            }
        }
        __syncthreads();
        // scores = q @ k^T (4 tiles, waves 0..3)
        for (int t = wav; t < 4; t += 8) {
            int m0 = (t & 1) * 16, n0 = (t >> 1) * 16;
            v8f acc = {};
#pragma unroll
            for (int ks = 0; ks < 4; ++ks) {
                v16bf aq = load_a_bf(s_q, HD, m0, ks * 32, lane);
                const bf16_t* p = s_k + (n0 + (lane & 15)) * HD + ks * 32 + (lane & 16);
                v16bf bb;
#pragma unroll
                for (int j = 0; j < 16; ++j) bb[j] = p[j];
                acc = wmma_bf(aq, bb, acc);
            }
            int cj = n0 + (lane & 15);
            int r0 = m0 + ((lane & 16) >> 1);
#pragma unroll
            for (int r = 0; r < 8; ++r) s_att[(r0 + r) * 32 + cj] = acc[r];
        }
        __syncthreads();
        // masked softmax over j (wave 0, one lane per row)
        if (tid < 32) {
            const float* mrow = masks + (size_t)b * NA * NA + tid * NA;
            float mx = -3.4e38f;
            for (int j = 0; j < 32; ++j) {
                float m = mrow[j];
                float sv = s_att[tid * 32 + j] * m - NEGC * (1.f - m);
                s_att[tid * 32 + j] = sv;
                mx = sv > mx ? sv : mx;
            }
            float ssum = 0.f;
            for (int j = 0; j < 32; ++j) {
                float e = __expf(s_att[tid * 32 + j] - mx);
                s_att[tid * 32 + j] = e; ssum += e;
            }
            float inv = 1.f / ssum;
            for (int j = 0; j < 32; ++j) s_att[tid * 32 + j] *= inv;
        }
        __syncthreads();
        // h = att @ v (16 tiles, K=32)
        for (int t = wav; t < 16; t += 8) {
            int m0 = (t & 1) * 16, n0 = (t >> 1) * 16;
            v16bf aa = load_a_f32(s_att, 32, m0, lane);
            v16bf bb;
            int n = n0 + (lane & 15), kb = lane & 16;
#pragma unroll
            for (int j = 0; j < 16; ++j) bb[j] = s_v[(kb + j) * HD + n];
            v8f acc = {};
            acc = wmma_bf(aa, bb, acc);
            int r0 = m0 + ((lane & 16) >> 1);
#pragma unroll
            for (int r = 0; r < 8; ++r) {
                int o = (r0 + r) * HD + n;
                s_h[o] = acc[r]; s_hb[o] = (bf16_t)acc[r];
            }
        }
        __syncthreads();
    }

    // ---- fc2 -> q output (waves 0,1) ----
    for (int t = wav; t < 2; t += 8) {
        int m0 = t * 16;
        v8f acc = {};
#pragma unroll
        for (int ks = 0; ks < 4; ++ks)
            acc = wmma_bf(load_a_bf(s_hb, HD, m0, ks * 32, lane),
                          load_b_ws(ws, WS_FC2, 4, 0, ks, lane), acc);
        int n = lane & 15;
        float bias = fc2_b[n];
        int r0 = m0 + ((lane & 16) >> 1);
#pragma unroll
        for (int r = 0; r < 8; ++r)
            out[(rowbase + r0 + r) * NACT + n] = acc[r] + bias;
    }
    // ---- h output ----
    const size_t qsz = (size_t)NB * NA * NACT;
    for (int i = tid; i < NA * HD; i += 256)
        out[qsz + rowbase * HD + i] = s_h[i];
}

extern "C" void kernel_launch(void* const* d_in, const int* in_sizes, int n_in,
                              void* d_out, int out_size, void* d_ws, size_t ws_size,
                              hipStream_t stream) {
    const float* inp    = (const float*)d_in[0];
    const float* hid    = (const float*)d_in[1];
    const float* masks  = (const float*)d_in[2];
    const float* fc1_w  = (const float*)d_in[3];
    const float* fc1_b  = (const float*)d_in[4];
    const float* w_ih   = (const float*)d_in[5];
    const float* b_ih   = (const float*)d_in[6];
    const float* w_hh   = (const float*)d_in[7];
    const float* b_hh   = (const float*)d_in[8];
    const float* fc2_w  = (const float*)d_in[9];
    const float* fc2_b  = (const float*)d_in[10];
    const float* a1v_w  = (const float*)d_in[11];
    const float* a1v_b  = (const float*)d_in[12];
    const float* a1k_w  = (const float*)d_in[13];
    const float* a1k_b  = (const float*)d_in[14];
    const float* a1q_w  = (const float*)d_in[15];
    const float* a1q_b  = (const float*)d_in[16];
    const float* a2v_w  = (const float*)d_in[17];
    const float* a2v_b  = (const float*)d_in[18];
    const float* a2k_w  = (const float*)d_in[19];
    const float* a2k_b  = (const float*)d_in[20];
    const float* a2q_w  = (const float*)d_in[21];
    const float* a2q_b  = (const float*)d_in[22];
    bf16_t* ws = (bf16_t*)d_ws;

    auto pk = [&](const float* W, int off, int ct, int ks, int K) {
        int total = ct * ks * 512;
        pack_w<<<(total + 255) / 256, 256, 0, stream>>>(W, ws + off, ct, ks, K);
    };
    pk(fc1_w, WS_FC1, 8, 3, 96);
    pk(w_ih,  WS_WIH, 24, 4, 128);
    pk(w_hh,  WS_WHH, 24, 4, 128);
    pk(a1v_w, WS_A1V, 8, 4, 128);
    pk(a1k_w, WS_A1K, 8, 4, 128);
    pk(a1q_w, WS_A1Q, 8, 4, 128);
    pk(a2v_w, WS_A2V, 8, 4, 128);
    pk(a2k_w, WS_A2K, 8, 4, 128);
    pk(a2q_w, WS_FC2 ? WS_A2Q : WS_A2Q, 8, 4, 128);
    pk(fc2_w, WS_FC2, 1, 4, 128);

    fused_agent<<<NB, 256, 0, stream>>>(
        inp, hid, masks, fc1_b, b_ih, b_hh, fc2_b,
        a1v_b, a1k_b, a1q_b, a2v_b, a2k_b, a2q_b,
        (const bf16_t*)d_ws, (float*)d_out);
}
